// LnSmoothStep_9964324127268
// MI455X (gfx1250) — matrix operand order repaired and checked by reading, not compiled, not run on hardware
//
#include <hip/hip_runtime.h>

typedef __attribute__((ext_vector_type(16))) _Float16 v16h;
typedef __attribute__((ext_vector_type(8)))  float    v8f;
typedef __attribute__((ext_vector_type(4)))  int      vi4;

constexpr int   kD   = 1024;
constexpr int   kN   = 8192;   // rows of x
constexpr int   kM   = 8192;   // rows of x_n
constexpr float kEps = 1e-12f;

constexpr int BM = 128, BN = 128, BK = 64;
constexpr int LDSU4   = 9;         // 72 halves per LDS row (64 data + 8 pad)
constexpr int KCHUNKS = kD / BK;   // 16

struct __align__(16) U4 { unsigned x, y, z, w; };
struct __align__(16) F4 { float x, y, z, w; };

union FragH {
  U4   u[2];
  v16h h;
};

#ifndef __has_builtin
#define __has_builtin(x) 0
#endif
#if __has_builtin(__builtin_amdgcn_global_load_async_to_lds_b128) && \
    __has_builtin(__builtin_amdgcn_s_wait_asynccnt)
#define USE_ASYNC_LDS 1
#else
#define USE_ASYNC_LDS 0
#endif

#if USE_ASYNC_LDS
__device__ __forceinline__ void cp_async16(void* lds_dst, const void* gsrc) {
  __builtin_amdgcn_global_load_async_to_lds_b128(
      (__attribute__((address_space(1))) vi4*)gsrc,
      (__attribute__((address_space(3))) vi4*)lds_dst,
      /*offset=*/0, /*cpol=*/0);
}
#endif

__device__ __forceinline__ unsigned pk2(float a, float b) {
  union { _Float16 h[2]; unsigned u; } q;
  q.h[0] = (_Float16)a;
  q.h[1] = (_Float16)b;
  return q.u;
}

// ---------------------------------------------------------------------------
// Kernel 0: f32 -> f16 pre-conversion (8 elements / thread).
// ---------------------------------------------------------------------------
__global__ __launch_bounds__(256)
void lnss_cvt(const float* __restrict__ src, _Float16* __restrict__ dst) {
  const size_t i = (size_t)blockIdx.x * blockDim.x + threadIdx.x;
  const F4* p = (const F4*)src + i * 2;
  const F4 a = p[0];
  const F4 b = p[1];
  U4 u{pk2(a.x, a.y), pk2(a.z, a.w), pk2(b.x, b.y), pk2(b.z, b.w)};
  ((U4*)dst)[i] = u;
}

// ---------------------------------------------------------------------------
// Kernel 1: per-row gate parameters  a, b, 1/n  (+ zero-init rowmax)
// One wave32 per row of x.
// ---------------------------------------------------------------------------
__global__ __launch_bounds__(256)
void lnss_params(const float* __restrict__ x,
                 const float* __restrict__ W,
                 const float* __restrict__ bias,
                 float* __restrict__ a_out,
                 float* __restrict__ b_out,
                 float* __restrict__ invn_out,
                 unsigned* __restrict__ rowmax) {
  const int row  = (int)((blockIdx.x * blockDim.x + threadIdx.x) >> 5);
  const int lane = threadIdx.x & 31;
  if (row >= kN) return;
  const float* xr = x + (size_t)row * kD;
  float s0 = 0.f, s1 = 0.f, s2 = 0.f;
  for (int k = lane; k < kD; k += 32) {
    const float xv = xr[k];
    s0 = fmaf(xv, W[k],          s0);
    s1 = fmaf(xv, W[kD + k],     s1);
    s2 = fmaf(xv, W[2 * kD + k], s2);
  }
  for (int off = 16; off > 0; off >>= 1) {
    s0 += __shfl_down(s0, off);
    s1 += __shfl_down(s1, off);
    s2 += __shfl_down(s2, off);
  }
  if (lane == 0) {
    const float lo = kEps, hi = 1.0f - kEps;
    const float p0 = fminf(fmaxf(1.f / (1.f + __expf(-(s0 + bias[0]))), lo), hi);
    const float p1 = fminf(fmaxf(1.f / (1.f + __expf(-(s1 + bias[1]))), lo), hi);
    const float p2 = fminf(fmaxf(1.f / (1.f + __expf(-(s2 + bias[2]))), lo), hi);
    a_out[row]    = p0;
    b_out[row]    = p0 + p1 * (1.f - p0);
    invn_out[row] = 1.f / p2;
    rowmax[row]   = 0u;   // sim_norm >= 0, so uint-bit atomicMax works
  }
}

// ---------------------------------------------------------------------------
// Kernel 2: sim_norm via f16 WMMA with async global->LDS pipeline + row max.
// 128x128 tile / block, 8 waves, each wave = 64x32 (4x2 WMMA tiles),
// BK=64 (two 16x16x32 k-steps per LDS buffer).
// ---------------------------------------------------------------------------
__global__ __launch_bounds__(256)
void lnss_gemm(const _Float16* __restrict__ xh,
               const _Float16* __restrict__ xnh,
               float* __restrict__ out,
               unsigned* __restrict__ rowmax) {
  __shared__ U4 As[2][BM * LDSU4];
  __shared__ U4 Bs[2][BN * LDSU4];

  const int tid   = threadIdx.x;
  const int lane  = tid & 31;
  const int wid   = tid >> 5;
  const int waveM = wid & 1;    // 64-row slice
  const int waveN = wid >> 1;   // 32-col slice
  const int l16   = lane & 15;
  const int lhi   = lane >> 4;  // 0 or 1

  const int rowBase = blockIdx.y * BM;
  const int colBase = blockIdx.x * BN;

  // Staging map: each wave copies 4 consecutive rows x 128B per j-step.
  const int srow = wid * 4 + (lane >> 3);  // 0..31, +32 per j
  const int scol = lane & 7;               // u4 column 0..7

  v8f acc[4][2];
  {
    v8f z;
    for (int e = 0; e < 8; ++e) z[e] = 0.f;
    for (int tm = 0; tm < 4; ++tm)
      for (int tn = 0; tn < 2; ++tn) acc[tm][tn] = z;
  }

#if USE_ASYNC_LDS
  auto stage = [&](int buf, int kc) {
#pragma unroll
    for (int j = 0; j < 4; ++j) {
      const int r = srow + j * 32;
      cp_async16(&As[buf][r * LDSU4 + scol],
                 xh + (size_t)(rowBase + r) * kD + kc * BK + scol * 8);
      cp_async16(&Bs[buf][r * LDSU4 + scol],
                 xnh + (size_t)(colBase + r) * kD + kc * BK + scol * 8);
    }
  };

  stage(0, 0);
  __builtin_amdgcn_s_wait_asynccnt(0);
  __syncthreads();
#else
  U4 ra[4], rb[4];
  auto gload = [&](int kc) {
#pragma unroll
    for (int j = 0; j < 4; ++j) {
      const int r = srow + j * 32;
      ra[j] = *(const U4*)(xh  + (size_t)(rowBase + r) * kD + kc * BK + scol * 8);
      rb[j] = *(const U4*)(xnh + (size_t)(colBase + r) * kD + kc * BK + scol * 8);
    }
  };
  auto lstore = [&](int buf) {
#pragma unroll
    for (int j = 0; j < 4; ++j) {
      const int r = srow + j * 32;
      As[buf][r * LDSU4 + scol] = ra[j];
      Bs[buf][r * LDSU4 + scol] = rb[j];
    }
  };

  gload(0);
  lstore(0);
  __syncthreads();
#endif

  for (int kc = 0; kc < KCHUNKS; ++kc) {
    const int cur = kc & 1;

#if USE_ASYNC_LDS
    if (kc + 1 < KCHUNKS) stage((kc + 1) & 1, kc + 1);  // DMA next buffer
#else
    if (kc + 1 < KCHUNKS) gload(kc + 1);
#endif

    // Two 16x16x32 k-steps inside this 64-wide chunk.
#pragma unroll
    for (int ks = 0; ks < 2; ++ks) {
      const int kb = ks * 4;  // u4 column base of this k-step
      FragH fa[4], fb[2];
#pragma unroll
      for (int tm = 0; tm < 4; ++tm) {
        const int r = (waveM * 64 + tm * 16 + l16) * LDSU4 + kb;
        fa[tm].u[0] = As[cur][r + lhi];      // K[0..7] / K[8..15]
        fa[tm].u[1] = As[cur][r + lhi + 2];  // K[16..23] / K[24..31]
      }
#pragma unroll
      for (int tn = 0; tn < 2; ++tn) {
        const int r = (waveN * 32 + tn * 16 + l16) * LDSU4 + kb;
        fb[tn].u[0] = Bs[cur][r + lhi];      // K[0..7] / K[8..15]
        fb[tn].u[1] = Bs[cur][r + lhi + 2];  // K[16..23] / K[24..31]
      }
#pragma unroll
      for (int tm = 0; tm < 4; ++tm)
#pragma unroll
        for (int tn = 0; tn < 2; ++tn)
          acc[tm][tn] = __builtin_amdgcn_wmma_f32_16x16x32_f16(
              false, fa[tm].h, false, fb[tn].h, (short)0, acc[tm][tn],
              false, false);
    }

#if USE_ASYNC_LDS
    if (kc + 1 < KCHUNKS) __builtin_amdgcn_s_wait_asynccnt(0);
#else
    if (kc + 1 < KCHUNKS) lstore((kc + 1) & 1);
#endif
    __syncthreads();
  }

  // Epilogue: sim_norm = dot * (1/(1024*2048)) + 0.5; store + fused row max.
  const float ksc = 1.0f / 2097152.0f;
#pragma unroll
  for (int tm = 0; tm < 4; ++tm)
#pragma unroll
    for (int tn = 0; tn < 2; ++tn)
#pragma unroll
      for (int e = 0; e < 8; ++e)
        acc[tm][tn][e] = fmaf(acc[tm][tn][e], ksc, 0.5f);

#pragma unroll
  for (int tm = 0; tm < 4; ++tm) {
#pragma unroll
    for (int e = 0; e < 8; ++e) {
      const int    r  = rowBase + waveM * 64 + tm * 16 + lhi * 8 + e;
      const size_t rb = (size_t)r * kM + colBase + waveN * 32 + l16;
      out[rb]      = acc[tm][0][e];
      out[rb + 16] = acc[tm][1][e];
      float m = fmaxf(acc[tm][0][e], acc[tm][1][e]);
      m = fmaxf(m, __shfl_xor(m, 1));
      m = fmaxf(m, __shfl_xor(m, 2));
      m = fmaxf(m, __shfl_xor(m, 4));
      m = fmaxf(m, __shfl_xor(m, 8));
      if (l16 == 0) atomicMax(rowmax + r, __float_as_uint(m));
    }
  }
}

// ---------------------------------------------------------------------------
// Kernel 3: in-place smooth-step scoring over d_out.
// ---------------------------------------------------------------------------
__global__ __launch_bounds__(256)
void lnss_score(float* __restrict__ out,
                const float* __restrict__ a_arr,
                const float* __restrict__ b_arr,
                const float* __restrict__ invn_arr,
                const unsigned* __restrict__ rowmax) {
  const size_t idx = ((size_t)blockIdx.x * blockDim.x + threadIdx.x) * 4;
  const int row = (int)(idx >> 13);  // 8192 cols per row, 4 elems per thread
  const float a    = fminf(a_arr[row], __uint_as_float(rowmax[row])) - kEps;
  const float b    = b_arr[row];
  const float invn = invn_arr[row];
  F4 s4 = *(const F4*)(out + idx);
  float sv[4] = {s4.x, s4.y, s4.z, s4.w};
  float rv[4];
#pragma unroll
  for (int e = 0; e < 4; ++e) {
    const float s   = sv[e];
    const float num = fmaxf(__expf(invn * __logf(fabsf(s - a))), kEps);
    const float den = num + fmaxf(__expf(invn * __logf(fabsf(b - s))), kEps);
    float sc = num / den;
    sc = (s < a) ? 0.f : sc;
    sc = (s > b) ? 1.f : sc;
    rv[e] = sc;
  }
  F4 o{rv[0], rv[1], rv[2], rv[3]};
  *(F4*)(out + idx) = o;
}

// ---------------------------------------------------------------------------
extern "C" void kernel_launch(void* const* d_in, const int* in_sizes, int n_in,
                              void* d_out, int out_size, void* d_ws, size_t ws_size,
                              hipStream_t stream) {
  (void)in_sizes; (void)n_in; (void)out_size; (void)ws_size;
  const float* x    = (const float*)d_in[0];
  const float* xn   = (const float*)d_in[1];
  const float* W    = (const float*)d_in[2];
  const float* bias = (const float*)d_in[3];
  float* out = (float*)d_out;

  // Workspace layout:
  //   xh   : 8192*1024 f16 (16 MB)
  //   xnh  : 8192*1024 f16 (16 MB)
  //   a, b, invn : 3 * 8192 f32
  //   rowmax     : 8192 u32           (total ~32.1 MB)
  _Float16* xh  = (_Float16*)d_ws;
  _Float16* xnh = xh + (size_t)kN * kD;
  float*    a_arr  = (float*)(xnh + (size_t)kM * kD);
  float*    b_arr  = a_arr + kN;
  float*    invn   = b_arr + kN;
  unsigned* rowmax = (unsigned*)(invn + kN);

  const unsigned cvt_blocks = (unsigned)((size_t)kN * kD / 8 / 256);
  lnss_cvt<<<cvt_blocks, 256, 0, stream>>>(x, xh);
  lnss_cvt<<<cvt_blocks, 256, 0, stream>>>(xn, xnh);

  lnss_params<<<kN / 8, 256, 0, stream>>>(x, W, bias, a_arr, b_arr, invn, rowmax);

  dim3 g(kM / BN, kN / BM);  // 64 x 64 tiles
  lnss_gemm<<<g, 256, 0, stream>>>(xh, xnh, out, rowmax);

  const size_t nthreads = (size_t)kN * kM / 4;
  lnss_score<<<(unsigned)(nthreads / 256), 256, 0, stream>>>(out, a_arr, b_arr,
                                                             invn, rowmax);
}
